// MultiHeadAttention_24326694765251
// MI455X (gfx1250) — matrix operand order repaired and checked by reading, not compiled, not run on hardware
//
#include <hip/hip_runtime.h>
#include <hip/hip_bf16.h>

// ---------------------------------------------------------------------------
// MHA forward on gfx1250 via V_WMMA_F32_16X16X32_BF16 (bf16 in, f32 accum).
// Outputs (concat in d_out): out [B,S,D] f32, attn [B,H,S,S] f32.
// ---------------------------------------------------------------------------

typedef __attribute__((ext_vector_type(16))) __bf16 v16bf;
typedef __attribute__((ext_vector_type(8)))  __bf16 v8bf;
typedef __attribute__((ext_vector_type(8)))  float  v8f;

constexpr int BB  = 2;
constexpr int SS  = 2048;
constexpr int DD  = 768;
constexpr int HH  = 12;
constexpr int DKK = 64;

// ---- WMMA core ------------------------------------------------------------
__device__ __forceinline__ v8f wmma_bf16(v16bf a, v16bf b, v8f c) {
  // (neg_a, A, neg_b, B, c_mod, C, reuse_a, reuse_b)
  return __builtin_amdgcn_wmma_f32_16x16x32_bf16(false, a, false, b,
                                                 (short)0, c, false, false);
}

// ---- Operand loaders (wave32, contiguous-K order shared by A and B) --------------------------
// A 16x32 (MxK) from row-major: lane l holds row M=l&15; 16 contiguous
// K-values per lane (elem e -> K=half*16+e).
__device__ __forceinline__ v16bf load_a_bf16(const __bf16* __restrict__ A,
                                             size_t lda, int lane) {
  const int half = lane >> 4;
  const int m    = lane & 15;
  const __bf16* p = A + (size_t)m * lda + half * 16;
  const v8bf lo = *(const v8bf*)(p);
  const v8bf hi = *(const v8bf*)(p + 8);
  v16bf r;
#pragma unroll
  for (int i = 0; i < 8; ++i) { r[i] = lo[i]; r[i + 8] = hi[i]; }
  return r;
}

__device__ __forceinline__ v16bf load_a_f32(const float* __restrict__ A,
                                            size_t lda, int lane) {
  const int half = lane >> 4;
  const int m    = lane & 15;
  const float* p = A + (size_t)m * lda + half * 16;
  v16bf r;
#pragma unroll
  for (int i = 0; i < 8; ++i) { r[i] = (__bf16)p[i]; r[i + 8] = (__bf16)p[i + 8]; }
  return r;
}

// B 32x16 (KxN) sourced from Bt = B^T stored row-major [N][K]:
// lane l: N=l&15, elems e -> K=(l>>4)*16+e => 16 contiguous K-values per lane.
__device__ __forceinline__ v16bf load_b_bf16(const __bf16* __restrict__ Bt,
                                             size_t ldb, int lane) {
  const int n  = lane & 15;
  const int kb = (lane >> 4) << 4;
  const __bf16* p = Bt + (size_t)n * ldb + kb;
  const v8bf lo = *(const v8bf*)(p);
  const v8bf hi = *(const v8bf*)(p + 8);
  v16bf r;
#pragma unroll
  for (int i = 0; i < 8; ++i) { r[i] = lo[i]; r[i + 8] = hi[i]; }
  return r;
}

__device__ __forceinline__ v16bf load_b_f32(const float* __restrict__ Bt,
                                            size_t ldb, int lane) {
  const int n  = lane & 15;
  const int kb = (lane >> 4) << 4;
  const float* p = Bt + (size_t)n * ldb + kb;
  v16bf r;
#pragma unroll
  for (int i = 0; i < 16; ++i) r[i] = (__bf16)p[i];
  return r;
}

// ---------------------------------------------------------------------------
// 1) projection: Y = X @ W^T + b.  W[out,in] row-major == [N][K] layout.
//    mode 0/1: bf16 [B,H,S,dk] ; mode 2: bf16 [B,H,dk,S] (v transposed)
// ---------------------------------------------------------------------------
__global__ __launch_bounds__(256)
void proj_kernel(const float* __restrict__ X, const float* __restrict__ W,
                 const float* __restrict__ bias, __bf16* __restrict__ Y, int mode) {
  const int lane = threadIdx.x & 31;
  const int wid  = (blockIdx.x * blockDim.x + threadIdx.x) >> 5;
  const int NT = DD / 16;                  // 48
  const int mt = wid / NT, nt = wid % NT;
  const int m0 = mt * 16, n0 = nt * 16;

  v8f acc = {};
#pragma unroll 4
  for (int k0 = 0; k0 < DD; k0 += 32) {
    v16bf a = load_a_f32(X + (size_t)m0 * DD + k0, DD, lane);
    v16bf b = load_b_f32(W + (size_t)n0 * DD + k0, DD, lane);
    acc = wmma_bf16(a, b, acc);
  }

  const int n  = n0 + (lane & 15);
  const int mb = m0 + ((lane >> 4) << 3);
  const int h = n >> 6, d = n & (DKK - 1);
#pragma unroll
  for (int r = 0; r < 8; ++r) {
    const int m  = mb + r;
    const int bI = m >> 11;
    const int s  = m & (SS - 1);
    const float val = acc[r] + bias[n];
    size_t idx;
    if (mode == 2) idx = (((size_t)bI * HH + h) * DKK + d) * SS + s;
    else           idx = (((size_t)bI * HH + h) * SS + s) * DKK + d;
    Y[idx] = (__bf16)val;
  }
}

// ---------------------------------------------------------------------------
// 2) scores[bh] = (q_bh @ k_bh^T) / sqrt(dk) -> f32 attn region of d_out
// ---------------------------------------------------------------------------
__global__ __launch_bounds__(256)
void scores_kernel(const __bf16* __restrict__ qh, const __bf16* __restrict__ kh,
                   float* __restrict__ attn) {
  const int lane = threadIdx.x & 31;
  const int wid  = (blockIdx.x * blockDim.x + threadIdx.x) >> 5;
  const int T = SS / 16;                   // 128
  const int perBH = T * T;                 // 16384
  const int bh  = wid / perBH;
  const int rem = wid % perBH;
  const int mt = rem / T, nt = rem % T;

  const __bf16* A  = qh + (size_t)bh * SS * DKK;
  const __bf16* Bt = kh + (size_t)bh * SS * DKK;

  v8f acc = {};
#pragma unroll
  for (int k0 = 0; k0 < DKK; k0 += 32) {
    v16bf a = load_a_bf16(A  + (size_t)(mt * 16) * DKK + k0, DKK, lane);
    v16bf b = load_b_bf16(Bt + (size_t)(nt * 16) * DKK + k0, DKK, lane);
    acc = wmma_bf16(a, b, acc);
  }

  float* C = attn + (size_t)bh * SS * SS;
  const int n  = nt * 16 + (lane & 15);
  const int mb = mt * 16 + ((lane >> 4) << 3);
#pragma unroll
  for (int r = 0; r < 8; ++r)
    C[(size_t)(mb + r) * SS + n] = acc[r] * 0.125f;   // 1/sqrt(64)
}

// ---------------------------------------------------------------------------
// 3) in-place row softmax (rows = B*H*S, cols = S = 2048), 256 thr/row
// ---------------------------------------------------------------------------
__global__ __launch_bounds__(256)
void softmax_kernel(float* __restrict__ attn) {
  __shared__ float red[256];
  float* p = attn + (size_t)blockIdx.x * SS;
  const int t = threadIdx.x;

  float v[8];
  float m = -3.402823466e38f;
#pragma unroll
  for (int i = 0; i < 8; ++i) { v[i] = p[t + i * 256]; m = fmaxf(m, v[i]); }
  red[t] = m; __syncthreads();
  for (int s = 128; s > 0; s >>= 1) {
    if (t < s) red[t] = fmaxf(red[t], red[t + s]);
    __syncthreads();
  }
  m = red[0]; __syncthreads();

  float sum = 0.f;
#pragma unroll
  for (int i = 0; i < 8; ++i) { v[i] = __expf(v[i] - m); sum += v[i]; }
  red[t] = sum; __syncthreads();
  for (int s = 128; s > 0; s >>= 1) {
    if (t < s) red[t] += red[t + s];
    __syncthreads();
  }
  const float inv = 1.0f / red[0];
#pragma unroll
  for (int i = 0; i < 8; ++i) p[t + i * 256] = v[i] * inv;
}

// ---------------------------------------------------------------------------
// 4) ctx[bh] = P_bh @ v_bh ; A = attn f32->bf16 inline, Bt = v^T [dk][S]
// ---------------------------------------------------------------------------
__global__ __launch_bounds__(256)
void ctx_kernel(const float* __restrict__ attn, const __bf16* __restrict__ vT,
                __bf16* __restrict__ ctx) {
  const int lane = threadIdx.x & 31;
  const int wid  = (blockIdx.x * blockDim.x + threadIdx.x) >> 5;
  const int NT = DKK / 16;                 // 4
  const int perBH = (SS / 16) * NT;        // 512
  const int bh  = wid / perBH;
  const int rem = wid % perBH;
  const int mt = rem / NT, nt = rem % NT;

  const float*  A  = attn + (size_t)bh * SS * SS;
  const __bf16* Bt = vT   + (size_t)bh * DKK * SS;

  v8f acc = {};
#pragma unroll 4
  for (int k0 = 0; k0 < SS; k0 += 32) {
    v16bf a = load_a_f32 (A  + (size_t)(mt * 16) * SS + k0, SS, lane);
    v16bf b = load_b_bf16(Bt + (size_t)(nt * 16) * SS + k0, SS, lane);
    acc = wmma_bf16(a, b, acc);
  }

  const int bI = bh / HH, h = bh % HH;
  const int n  = nt * 16 + (lane & 15);
  const int mb = mt * 16 + ((lane >> 4) << 3);
#pragma unroll
  for (int r = 0; r < 8; ++r) {
    const int s = mb + r;
    ctx[((size_t)bI * SS + s) * DD + h * DKK + n] = (__bf16)acc[r];
  }
}

// ---------------------------------------------------------------------------
// 5) out = ctx @ Wo^T + bo -> f32 d_out[0 .. B*S*D)
// ---------------------------------------------------------------------------
__global__ __launch_bounds__(256)
void oproj_kernel(const __bf16* __restrict__ ctx, const float* __restrict__ Wo,
                  const float* __restrict__ bo, float* __restrict__ out) {
  const int lane = threadIdx.x & 31;
  const int wid  = (blockIdx.x * blockDim.x + threadIdx.x) >> 5;
  const int NT = DD / 16;
  const int mt = wid / NT, nt = wid % NT;

  v8f acc = {};
#pragma unroll 4
  for (int k0 = 0; k0 < DD; k0 += 32) {
    v16bf a = load_a_bf16(ctx + (size_t)(mt * 16) * DD + k0, DD, lane);
    v16bf b = load_b_f32 (Wo  + (size_t)(nt * 16) * DD + k0, DD, lane);
    acc = wmma_bf16(a, b, acc);
  }

  const int n  = nt * 16 + (lane & 15);
  const int mb = mt * 16 + ((lane >> 4) << 3);
#pragma unroll
  for (int r = 0; r < 8; ++r)
    out[(size_t)(mb + r) * DD + n] = acc[r] + bo[n];
}

// ---------------------------------------------------------------------------
extern "C" void kernel_launch(void* const* d_in, const int* in_sizes, int n_in,
                              void* d_out, int out_size, void* d_ws, size_t ws_size,
                              hipStream_t stream) {
  const float* Q  = (const float*)d_in[0];
  const float* K  = (const float*)d_in[1];
  const float* V  = (const float*)d_in[2];
  const float* Wq = (const float*)d_in[3];
  const float* bq = (const float*)d_in[4];
  const float* Wk = (const float*)d_in[5];
  const float* bk = (const float*)d_in[6];
  const float* Wv = (const float*)d_in[7];
  const float* bv = (const float*)d_in[8];
  const float* Wo = (const float*)d_in[9];
  const float* bo = (const float*)d_in[10];

  float* out  = (float*)d_out;                       // [B,S,D]
  float* attn = out + (size_t)BB * SS * DD;          // [B,H,S,S]

  const size_t nTok = (size_t)BB * SS * DD;          // 3,145,728 elems
  char* w = (char*)d_ws;                             // bf16 scratch (~25 MB)
  __bf16* qh   = (__bf16*)(w + 0 * nTok * 2);        // [B,H,S,dk]
  __bf16* kh   = (__bf16*)(w + 1 * nTok * 2);        // [B,H,S,dk]
  __bf16* vT   = (__bf16*)(w + 2 * nTok * 2);        // [B,H,dk,S]
  __bf16* ctxb = (__bf16*)(w + 3 * nTok * 2);        // [B,S,D]

  // wave counts: proj/oproj 256*48=12288, scores 24*16384=393216,
  // ctx 24*512=12288 -> all divisible by 8 waves/block (EXEC all-ones).
  proj_kernel   <<<1536,        256, 0, stream>>>(Q, Wq, bq, qh, 0);
  proj_kernel   <<<1536,        256, 0, stream>>>(K, Wk, bk, kh, 1);
  proj_kernel   <<<1536,        256, 0, stream>>>(V, Wv, bv, vT, 2);
  scores_kernel <<<49152,       256, 0, stream>>>(qh, kh, attn);
  softmax_kernel<<<BB * HH * SS,256, 0, stream>>>(attn);
  ctx_kernel    <<<1536,        256, 0, stream>>>(attn, vT, ctxb);
  oproj_kernel  <<<1536,        256, 0, stream>>>(ctxb, Wo, bo, out);
}
